// DoubleLayerTextLSTM_9311489098011
// MI455X (gfx1250) — compile-verified, matrix operand order checked by reading, not run on hardware
//
#include <hip/hip_runtime.h>

// ---------------------------------------------------------------------------
// Types for CDNA5 WMMA (wave32): v_wmma_f32_16x16x32_bf16
// ---------------------------------------------------------------------------
typedef __attribute__((ext_vector_type(16))) __bf16       v16bf;
typedef __attribute__((ext_vector_type(8)))  float        v8f;
typedef __attribute__((ext_vector_type(8)))  unsigned int v8u;

#define WMMA_BF16(a, b, c) \
  __builtin_amdgcn_wmma_f32_16x16x32_bf16(false, (a), false, (b), (short)0, (c), false, false)

__device__ __forceinline__ unsigned short f2bf(float x) {
  unsigned int u = __float_as_uint(x);
  u += 0x7FFFu + ((u >> 16) & 1u);          // round-to-nearest-even
  return (unsigned short)(u >> 16);
}
__device__ __forceinline__ float bf2f(unsigned short h) {
  return __uint_as_float(((unsigned int)h) << 16);
}
__device__ __forceinline__ float sigmoidf_(float x) { return 1.0f / (1.0f + __expf(-x)); }
__device__ __forceinline__ float tanhf_(float x) {
  float t = __expf(-2.0f * fabsf(x));
  float r = (1.0f - t) / (1.0f + t);
  return copysignf(r, x);
}

// Load one WMMA 16-bit A/B fragment for this lane from a row-major [*,K] matrix.
// Per ISA 7.12.2 (16-bit A 16x32): lanes 0-15 hold K{0..7}+{16..23} of row (lane&15),
// lanes 16-31 hold K{8..15}+{24..31}. Caller passes p = row_base + k0 + ((lane&16)?8:0).
__device__ __forceinline__ v16bf load_frag(const unsigned short* p) {
  uint4 lo = *reinterpret_cast<const uint4*>(p);
  uint4 hi = *reinterpret_cast<const uint4*>(p + 16);
  v8u u;
  u[0] = lo.x; u[1] = lo.y; u[2] = lo.z; u[3] = lo.w;
  u[4] = hi.x; u[5] = hi.y; u[6] = hi.z; u[7] = hi.w;
  return __builtin_bit_cast(v16bf, u);
}

// ---------------------------------------------------------------------------
// One-time (per launch) data prep kernels
// ---------------------------------------------------------------------------

// dst[n*K + k] = bf16(src[k*N + n])   (weight transpose + f32->bf16)
__global__ void transpose_bf16_kernel(const float* __restrict__ src,
                                      unsigned short* __restrict__ dst,
                                      int K, int N) {
  size_t idx = (size_t)blockIdx.x * blockDim.x + threadIdx.x;
  size_t total = (size_t)K * N;
  if (idx >= total) return;
  int    k = (int)(idx % (size_t)K);
  size_t n = idx / (size_t)K;
  dst[idx] = f2bf(src[(size_t)k * N + n]);
}

// Time-major embedding gather: Xe[t*64 + b][e] = bf16(E[X[b][t]][e])
__global__ void embed_kernel(const int* __restrict__ X, const float* __restrict__ E,
                             unsigned short* __restrict__ Xe) {
  size_t idx = (size_t)blockIdx.x * blockDim.x + threadIdx.x;
  constexpr size_t total = (size_t)256 * 64 * 1024;
  if (idx >= total) return;
  int    e   = (int)(idx & 1023);
  size_t row = idx >> 10;
  int    b   = (int)(row & 63);
  int    t   = (int)(row >> 6);
  int    tok = X[b * 256 + t];
  Xe[idx] = f2bf(E[(size_t)tok * 1024 + e]);
}

__global__ void init_state_kernel(const float* __restrict__ H, const float* __restrict__ C,
                                  unsigned short* __restrict__ H1, unsigned short* __restrict__ H2,
                                  float* __restrict__ Cs, float* __restrict__ Cl2) {
  int i = blockIdx.x * blockDim.x + threadIdx.x;
  if (i >= 64 * 1024) return;
  unsigned short hb = f2bf(H[i]);
  H1[i] = hb; H2[i] = hb;          // scan init carry = (H, C, H, C)
  Cs[i] = C[i]; Cl2[i] = C[i];
}

// ---------------------------------------------------------------------------
// Big parallel GEMM: D_bf16[M,N] = A_bf16[M,K] x Bt_bf16[N,K]^T
// One wave -> 16(M) x 64(N) tile. Double-buffered (ping/pong) K-loop, unrolled
// by 2 so loads land directly in the register set they are consumed from (no
// rotation copies), while the other set's WMMAs execute under partial waits.
// ---------------------------------------------------------------------------
__global__ void __launch_bounds__(128)
gemm_bf16_n64_kernel(const unsigned short* __restrict__ A,
                     const unsigned short* __restrict__ Bt,
                     unsigned short* __restrict__ D,
                     int M, int N, int K) {
  const int lane  = threadIdx.x & 31;
  const int wid   = blockIdx.x * 4 + (threadIdx.x >> 5);
  const int ntile = N >> 6;
  const int mt    = wid / ntile;
  const int nt    = wid - mt * ntile;
  if (mt * 16 >= M) return;

  const int kbase = (lane & 16) ? 8 : 0;
  const int bl    = lane & 15;
  const unsigned short* Ar = A + (size_t)(mt * 16 + bl) * K + kbase;
  const unsigned short* B0 = Bt + (size_t)(nt * 64 +  0 + bl) * K + kbase;
  const unsigned short* B1 = Bt + (size_t)(nt * 64 + 16 + bl) * K + kbase;
  const unsigned short* B2 = Bt + (size_t)(nt * 64 + 32 + bl) * K + kbase;
  const unsigned short* B3 = Bt + (size_t)(nt * 64 + 48 + bl) * K + kbase;

  v8f acc0 = {}, acc1 = {}, acc2 = {}, acc3 = {};

  // Ping-pong fragment sets.
  v16bf aP, p0, p1, p2, p3;   // set P
  v16bf aQ, q0, q1, q2, q3;   // set Q

  aP = load_frag(Ar); p0 = load_frag(B0); p1 = load_frag(B1);
  p2 = load_frag(B2); p3 = load_frag(B3);

#pragma unroll 1
  for (int k0 = 0; k0 < K - 64; k0 += 64) {
    const int k1 = k0 + 32;
    aQ = load_frag(Ar + k1); q0 = load_frag(B0 + k1); q1 = load_frag(B1 + k1);
    q2 = load_frag(B2 + k1); q3 = load_frag(B3 + k1);
    acc0 = WMMA_BF16(aP, p0, acc0);
    acc1 = WMMA_BF16(aP, p1, acc1);
    acc2 = WMMA_BF16(aP, p2, acc2);
    acc3 = WMMA_BF16(aP, p3, acc3);
    const int k2 = k0 + 64;
    aP = load_frag(Ar + k2); p0 = load_frag(B0 + k2); p1 = load_frag(B1 + k2);
    p2 = load_frag(B2 + k2); p3 = load_frag(B3 + k2);
    acc0 = WMMA_BF16(aQ, q0, acc0);
    acc1 = WMMA_BF16(aQ, q1, acc1);
    acc2 = WMMA_BF16(aQ, q2, acc2);
    acc3 = WMMA_BF16(aQ, q3, acc3);
  }
  {
    const int k1 = K - 32;
    aQ = load_frag(Ar + k1); q0 = load_frag(B0 + k1); q1 = load_frag(B1 + k1);
    q2 = load_frag(B2 + k1); q3 = load_frag(B3 + k1);
    acc0 = WMMA_BF16(aP, p0, acc0);
    acc1 = WMMA_BF16(aP, p1, acc1);
    acc2 = WMMA_BF16(aP, p2, acc2);
    acc3 = WMMA_BF16(aP, p3, acc3);
    acc0 = WMMA_BF16(aQ, q0, acc0);
    acc1 = WMMA_BF16(aQ, q1, acc1);
    acc2 = WMMA_BF16(aQ, q2, acc2);
    acc3 = WMMA_BF16(aQ, q3, acc3);
  }

  // C/D layout: VGPR r -> M = r (lanes 0-15) or r+8 (lanes 16-31), N = lane&15.
  const int mrb = mt * 16 + ((lane & 16) ? 8 : 0);
  const int c0  = nt * 64 + bl;
  for (int r = 0; r < 8; ++r) {
    size_t row = (size_t)(mrb + r) * N;
    D[row + c0 +  0] = f2bf(acc0[r]);
    D[row + c0 + 16] = f2bf(acc1[r]);
    D[row + c0 + 32] = f2bf(acc2[r]);
    D[row + c0 + 48] = f2bf(acc3[r]);
  }
}

// ---------------------------------------------------------------------------
// One recurrent half-step (used for both LSTM layers):
//   gates[g] = A @ WhT[g] + xproj[g] + bias[g]   (g = f,i,c~,o ; gate o uses A3)
//   Hout = sig(o) * tanh(Cold) ; Cst = sig(f)*Cold + sig(i)*tanh(c~)   (in place)
// One wave computes the SAME 16x16 (batch,hid) tile for all 4 gates, so the
// cell update fuses entirely in registers. 256 waves -> latency-critical, so
// the K-loop is ping-pong double-buffered. SAME_A=true (layer 1) skips A3.
// ---------------------------------------------------------------------------
template <bool SAME_A>
__global__ void __launch_bounds__(128)
lstm_step_kernel(const unsigned short* __restrict__ A012,  // H source for gates f,i,c
                 const unsigned short* __restrict__ A3,    // H source for gate o
                 const unsigned short* __restrict__ WhT,   // [4096,1024] bf16
                 const unsigned short* __restrict__ xp,    // Xproj + t*64*8192 (+4096 for L2), row stride 8192
                 const float* __restrict__ b0, const float* __restrict__ b1,
                 const float* __restrict__ b2, const float* __restrict__ b3,
                 float* __restrict__ Cst, unsigned short* __restrict__ Hout) {
  constexpr int K = 1024;
  const int lane  = threadIdx.x & 31;
  const int wid   = blockIdx.x * 4 + (threadIdx.x >> 5);
  const int mt    = wid >> 6;   // 0..3   (batch tile)
  const int ht    = wid & 63;   // 0..63  (hid tile)
  const int kbase = (lane & 16) ? 8 : 0;
  const int bl    = lane & 15;
  const int h0    = ht * 16;

  const unsigned short* Ar  = A012 + (size_t)(mt * 16 + bl) * K + kbase;
  const unsigned short* Ar3 = A3   + (size_t)(mt * 16 + bl) * K + kbase;
  const unsigned short* Bf  = WhT + (size_t)(       h0 + bl) * K + kbase;
  const unsigned short* Bi  = WhT + (size_t)(1024 + h0 + bl) * K + kbase;
  const unsigned short* Bc  = WhT + (size_t)(2048 + h0 + bl) * K + kbase;
  const unsigned short* Bo  = WhT + (size_t)(3072 + h0 + bl) * K + kbase;

  v8f aF = {}, aI = {}, aC = {}, aO = {};

  v16bf aP, a3P, fP, iP, cP, oP;   // set P
  v16bf aQ, a3Q, fQ, iQ, cQ, oQ;   // set Q

  aP = load_frag(Ar);
  a3P = SAME_A ? aP : load_frag(Ar3);
  fP = load_frag(Bf); iP = load_frag(Bi); cP = load_frag(Bc); oP = load_frag(Bo);

#pragma unroll 1
  for (int k0 = 0; k0 < K - 64; k0 += 64) {
    const int k1 = k0 + 32;
    aQ = load_frag(Ar + k1);
    a3Q = SAME_A ? aQ : load_frag(Ar3 + k1);
    fQ = load_frag(Bf + k1); iQ = load_frag(Bi + k1);
    cQ = load_frag(Bc + k1); oQ = load_frag(Bo + k1);
    aF = WMMA_BF16(aP,  fP, aF);
    aI = WMMA_BF16(aP,  iP, aI);
    aC = WMMA_BF16(aP,  cP, aC);
    aO = WMMA_BF16(a3P, oP, aO);
    const int k2 = k0 + 64;
    aP = load_frag(Ar + k2);
    a3P = SAME_A ? aP : load_frag(Ar3 + k2);
    fP = load_frag(Bf + k2); iP = load_frag(Bi + k2);
    cP = load_frag(Bc + k2); oP = load_frag(Bo + k2);
    aF = WMMA_BF16(aQ,  fQ, aF);
    aI = WMMA_BF16(aQ,  iQ, aI);
    aC = WMMA_BF16(aQ,  cQ, aC);
    aO = WMMA_BF16(a3Q, oQ, aO);
  }
  {
    const int k1 = K - 32;
    aQ = load_frag(Ar + k1);
    a3Q = SAME_A ? aQ : load_frag(Ar3 + k1);
    fQ = load_frag(Bf + k1); iQ = load_frag(Bi + k1);
    cQ = load_frag(Bc + k1); oQ = load_frag(Bo + k1);
    aF = WMMA_BF16(aP,  fP, aF);
    aI = WMMA_BF16(aP,  iP, aI);
    aC = WMMA_BF16(aP,  cP, aC);
    aO = WMMA_BF16(a3P, oP, aO);
    aF = WMMA_BF16(aQ,  fQ, aF);
    aI = WMMA_BF16(aQ,  iQ, aI);
    aC = WMMA_BF16(aQ,  cQ, aC);
    aO = WMMA_BF16(a3Q, oQ, aO);
  }

  const int mrb = mt * 16 + ((lane & 16) ? 8 : 0);
  const int h   = h0 + bl;
  const float vb0 = b0[h], vb1 = b1[h], vb2 = b2[h], vb3 = b3[h];
  for (int r = 0; r < 8; ++r) {
    const int    m    = mrb + r;
    const size_t xrow = (size_t)m * 8192;
    float gf = aF[r] + bf2f(xp[xrow +        h]) + vb0;
    float gi = aI[r] + bf2f(xp[xrow + 1024 + h]) + vb1;
    float gc = aC[r] + bf2f(xp[xrow + 2048 + h]) + vb2;
    float go = aO[r] + bf2f(xp[xrow + 3072 + h]) + vb3;
    float f  = sigmoidf_(gf);
    float ii = sigmoidf_(gi);
    float ct = tanhf_(gc);
    float o  = sigmoidf_(go);
    const size_t s    = (size_t)m * 1024 + h;
    float        cold = Cst[s];
    Hout[s] = f2bf(o * tanhf_(cold));     // reference uses OLD cell state for H
    Cst[s]  = f * cold + ii * ct;
  }
}

// ---------------------------------------------------------------------------
// Logits: Out_f32[64,32000] = Hl2_bf16[64,1024] x WhqT_bf16[32000,1024]^T + bq
// ---------------------------------------------------------------------------
__global__ void __launch_bounds__(128)
logits_gemm_kernel(const unsigned short* __restrict__ A,
                   const unsigned short* __restrict__ Bt,
                   const float* __restrict__ bias,
                   float* __restrict__ Out) {
  constexpr int K = 1024, N = 32000;
  const int lane  = threadIdx.x & 31;
  const int wid   = blockIdx.x * 4 + (threadIdx.x >> 5);  // 2000 waves
  const int mt    = wid / 500;
  const int nt    = wid - mt * 500;
  const int kbase = (lane & 16) ? 8 : 0;
  const int bl    = lane & 15;

  const unsigned short* Ar = A + (size_t)(mt * 16 + bl) * K + kbase;
  const unsigned short* B0 = Bt + (size_t)(nt * 64 +  0 + bl) * K + kbase;
  const unsigned short* B1 = Bt + (size_t)(nt * 64 + 16 + bl) * K + kbase;
  const unsigned short* B2 = Bt + (size_t)(nt * 64 + 32 + bl) * K + kbase;
  const unsigned short* B3 = Bt + (size_t)(nt * 64 + 48 + bl) * K + kbase;

  v8f acc0 = {}, acc1 = {}, acc2 = {}, acc3 = {};

  v16bf aP, p0, p1, p2, p3;
  v16bf aQ, q0, q1, q2, q3;

  aP = load_frag(Ar); p0 = load_frag(B0); p1 = load_frag(B1);
  p2 = load_frag(B2); p3 = load_frag(B3);

#pragma unroll 1
  for (int k0 = 0; k0 < K - 64; k0 += 64) {
    const int k1 = k0 + 32;
    aQ = load_frag(Ar + k1); q0 = load_frag(B0 + k1); q1 = load_frag(B1 + k1);
    q2 = load_frag(B2 + k1); q3 = load_frag(B3 + k1);
    acc0 = WMMA_BF16(aP, p0, acc0);
    acc1 = WMMA_BF16(aP, p1, acc1);
    acc2 = WMMA_BF16(aP, p2, acc2);
    acc3 = WMMA_BF16(aP, p3, acc3);
    const int k2 = k0 + 64;
    aP = load_frag(Ar + k2); p0 = load_frag(B0 + k2); p1 = load_frag(B1 + k2);
    p2 = load_frag(B2 + k2); p3 = load_frag(B3 + k2);
    acc0 = WMMA_BF16(aQ, q0, acc0);
    acc1 = WMMA_BF16(aQ, q1, acc1);
    acc2 = WMMA_BF16(aQ, q2, acc2);
    acc3 = WMMA_BF16(aQ, q3, acc3);
  }
  {
    const int k1 = K - 32;
    aQ = load_frag(Ar + k1); q0 = load_frag(B0 + k1); q1 = load_frag(B1 + k1);
    q2 = load_frag(B2 + k1); q3 = load_frag(B3 + k1);
    acc0 = WMMA_BF16(aP, p0, acc0);
    acc1 = WMMA_BF16(aP, p1, acc1);
    acc2 = WMMA_BF16(aP, p2, acc2);
    acc3 = WMMA_BF16(aP, p3, acc3);
    acc0 = WMMA_BF16(aQ, q0, acc0);
    acc1 = WMMA_BF16(aQ, q1, acc1);
    acc2 = WMMA_BF16(aQ, q2, acc2);
    acc3 = WMMA_BF16(aQ, q3, acc3);
  }

  const int mrb = mt * 16 + ((lane & 16) ? 8 : 0);
  const int c0  = nt * 64 + bl;
  for (int r = 0; r < 8; ++r) {
    size_t row = (size_t)(mrb + r) * N;
    Out[row + c0 +  0] = acc0[r] + bias[c0 +  0];
    Out[row + c0 + 16] = acc1[r] + bias[c0 + 16];
    Out[row + c0 + 32] = acc2[r] + bias[c0 + 32];
    Out[row + c0 + 48] = acc3[r] + bias[c0 + 48];
  }
}

// ---------------------------------------------------------------------------
// Host orchestration
// ---------------------------------------------------------------------------
extern "C" void kernel_launch(void* const* d_in, const int* in_sizes, int n_in,
                              void* d_out, int out_size, void* d_ws, size_t ws_size,
                              hipStream_t stream) {
  constexpr int EMB = 1024, HID = 1024, BATCH = 64, TLEN = 256, NCLS = 32000;
  (void)in_sizes; (void)n_in; (void)out_size;

  // Input indices per setup_inputs() dict order:
  // 0:X 1:H 2:C 3:E 4:Wxf 5:Whf 6:Wxi 7:Whi 8:Wxc 9:Whc 10:Wxo 11:Who
  // 12:Wxfl2 13:Whfl2 14:Wxil2 15:Whil2 16:Wxcl2 17:Whcl2 18:Wxol2 19:Whol2
  // 20:Wxn 21:Wnn (dead) 22:bf 23:bi 24:bc 25:bo 26:bfl2 27:bil2 28:bcl2 29:bol2
  // 30:bn (dead) 31:Whq 32:bq

  char*  ws  = (char*)d_ws;
  size_t off = 0;
  auto carve = [&](size_t bytes) -> char* {
    char* p = ws + off;
    off = (off + bytes + 255) & ~(size_t)255;
    return p;
  };
  unsigned short* WxAllT = (unsigned short*)carve((size_t)8 * HID * EMB * 2);    // [8192,1024]
  unsigned short* Wh1T   = (unsigned short*)carve((size_t)4 * HID * HID * 2);    // [4096,1024]
  unsigned short* Wh2T   = (unsigned short*)carve((size_t)4 * HID * HID * 2);    // [4096,1024]
  unsigned short* WhqT   = (unsigned short*)carve((size_t)NCLS * HID * 2);       // [32000,1024]
  unsigned short* XeBf   = (unsigned short*)carve((size_t)TLEN * BATCH * EMB * 2);
  unsigned short* Xproj  = (unsigned short*)carve((size_t)TLEN * BATCH * 8 * HID * 2); // [16384,8192]
  unsigned short* H1a    = (unsigned short*)carve((size_t)BATCH * HID * 2);
  unsigned short* H1b    = (unsigned short*)carve((size_t)BATCH * HID * 2);
  unsigned short* H2a    = (unsigned short*)carve((size_t)BATCH * HID * 2);
  unsigned short* H2b    = (unsigned short*)carve((size_t)BATCH * HID * 2);
  float*          Cst    = (float*)carve((size_t)BATCH * HID * 4);
  float*          Cl2    = (float*)carve((size_t)BATCH * HID * 4);
  if (off > ws_size) return;  // workspace too small; bail deterministically

  // 1) Transpose + convert all weights to bf16 [N,K] so WMMA B-operand loads
  //    use the same contiguous-16B fragment addressing as the A-operand.
  auto t2bf = [&](int idx, unsigned short* dst, int K, int N) {
    size_t total = (size_t)K * N;
    int blocks = (int)((total + 255) / 256);
    transpose_bf16_kernel<<<blocks, 256, 0, stream>>>((const float*)d_in[idx], dst, K, N);
  };
  const int xw[8]  = {4, 6, 8, 10, 12, 14, 16, 18};   // Wxf,Wxi,Wxc,Wxo,Wxfl2,Wxil2,Wxcl2,Wxol2
  const int h1w[4] = {5, 7, 9, 11};                   // Whf,Whi,Whc,Who
  const int h2w[4] = {13, 15, 17, 19};                // Whfl2,Whil2,Whcl2,Whol2
  for (int g = 0; g < 8; ++g) t2bf(xw[g],  WxAllT + (size_t)g * HID * EMB, EMB, HID);
  for (int g = 0; g < 4; ++g) t2bf(h1w[g], Wh1T   + (size_t)g * HID * HID, HID, HID);
  for (int g = 0; g < 4; ++g) t2bf(h2w[g], Wh2T   + (size_t)g * HID * HID, HID, HID);
  t2bf(31, WhqT, HID, NCLS);

  // 2) Embedding gather (time-major, bf16) + state init.
  {
    size_t total = (size_t)TLEN * BATCH * EMB;
    embed_kernel<<<(int)((total + 255) / 256), 256, 0, stream>>>(
        (const int*)d_in[0], (const float*)d_in[3], XeBf);
  }
  init_state_kernel<<<(BATCH * HID + 255) / 256, 256, 0, stream>>>(
      (const float*)d_in[1], (const float*)d_in[2], H1a, H2a, Cst, Cl2);

  // 3) Hoisted input projections for ALL timesteps and ALL 8 gates:
  //    Xproj[16384, 8192] = XeBf[16384,1024] x WxAllT[8192,1024]^T
  {
    int blocks = (16384 / 16) * (8192 / 64) / 4;  // 32768 blocks x 4 waves
    gemm_bf16_n64_kernel<<<blocks, 128, 0, stream>>>(XeBf, WxAllT, Xproj, 16384, 8192, 1024);
  }

  // 4) Sequential recurrence: 2 dependent fused GEMM+cell-update kernels/step.
  //    Wh weights (16 MB bf16) stay L2-resident across all 256 steps.
  unsigned short* h1[2] = {H1a, H1b};
  unsigned short* h2[2] = {H2a, H2b};
  int c1 = 0, c2 = 0;
  for (int t = 0; t < TLEN; ++t) {
    const unsigned short* xp = Xproj + (size_t)t * BATCH * 8192;
    // Layer 1: all gates read old H.
    lstm_step_kernel<true><<<64, 128, 0, stream>>>(
        h1[c1], h1[c1], Wh1T, xp,
        (const float*)d_in[22], (const float*)d_in[23],
        (const float*)d_in[24], (const float*)d_in[25],
        Cst, h1[c1 ^ 1]);
    // Layer 2: F2/I2/C2 read new Hn; O2 reads old Hl2.
    lstm_step_kernel<false><<<64, 128, 0, stream>>>(
        h1[c1 ^ 1], h2[c2], Wh2T, xp + 4096,
        (const float*)d_in[26], (const float*)d_in[27],
        (const float*)d_in[28], (const float*)d_in[29],
        Cl2, h2[c2 ^ 1]);
    c1 ^= 1; c2 ^= 1;
  }

  // 5) Vocab projection: d_out[64,32000] = Hl2 x Whq + bq
  logits_gemm_kernel<<<500, 128, 0, stream>>>(h2[c2], WhqT, (const float*)d_in[32], (float*)d_out);
}
